// BiMambaBlock_85727547228498
// MI455X (gfx1250) — matrix operand order repaired and checked by reading, not compiled, not run on hardware
//
#include <hip/hip_runtime.h>
#include <hip/hip_bf16.h>

// ---------------- problem constants ----------------
#define DMODEL  256
#define DI      512          // d_inner
#define NSTATE  16
#define RANK    16
#define LSEQ    1024
#define BATCH   4
#define MROWS   (BATCH * LSEQ)   // 4096
#define NCHUNK  8
#define LCHUNK  (LSEQ / NCHUNK)  // 128

typedef __attribute__((ext_vector_type(16))) __bf16 v16bf;
typedef __attribute__((ext_vector_type(8)))  __bf16 v8bf;
typedef __attribute__((ext_vector_type(4)))  __bf16 v4bf;
typedef __attribute__((ext_vector_type(8)))  float  v8f;

// RNE float -> bf16 (only in bandwidth-bound converter/epilogue paths)
static __device__ __forceinline__ __bf16 f2bf(float f) {
    unsigned u = __float_as_uint(f);
    u += 0x7FFFu + ((u >> 16) & 1u);
    unsigned short h = (unsigned short)(u >> 16);
    return __builtin_bit_cast(__bf16, h);
}

// A fragment (16x32 bf16, MxK): lane m = lane&15; lane>=16 -> K offset +8.
// elements 0..7 -> K = k0+hi+e ; elements 8..15 -> K = k0+hi+16+(e-8)
static __device__ __forceinline__ v16bf load_frag_a(const __bf16* __restrict__ row,
                                                    int k0, int hi) {
    v8bf lo = *(const v8bf*)(row + k0 + hi);
    v8bf hb = *(const v8bf*)(row + k0 + hi + 16);
    v16bf v;
#pragma unroll
    for (int e = 0; e < 8; ++e) { v[e] = lo[e]; v[8 + e] = hb[e]; }
    return v;
}

// B fragment (32x16 bf16, KxN): lane n = lane&15; lane>=16 -> K offset +8.
// elements 0..7 -> K = k0+khalf+e ; elements 8..15 -> K = k0+khalf+16+(e-8)
static __device__ __forceinline__ v16bf load_frag_b(const __bf16* __restrict__ row,
                                                    int k0, int khalf) {
    v8bf lo = *(const v8bf*)(row + k0 + khalf);
    v8bf hb = *(const v8bf*)(row + k0 + khalf + 16);
    v16bf v;
#pragma unroll
    for (int e = 0; e < 8; ++e) { v[e] = lo[e]; v[8 + e] = hb[e]; }
    return v;
}

// Generic GEMM: out[m, n] = sum_k act[m,k] * W[n,k]  (+bias[n])
// One wave computes a 16 x (16*NT) tile with a copy-free ping-pong pipeline
// (requires K % 64 == 0): loads of one buffer set overlap wmmas of the other.
// flipIn:  act row m reads source row b*L + (L-1-l)
// flipOut: output row m writes dest   row b*L + (L-1-l)
// OBF:     write bf16 output (else f32)
template <int NT, bool OBF>
__global__ void gemm_bf16_wmma(const __bf16* __restrict__ act,
                               const __bf16* __restrict__ W,
                               const float* __restrict__ bias,
                               void* __restrict__ outp,
                               int M, int N, int K,
                               int flipIn, int flipOut,
                               int outStride, int outColOff) {
    const int lane = threadIdx.x & 31;
    const int wave = threadIdx.x >> 5;
    const int mtiles = M >> 4;
    const int unit = blockIdx.x * (blockDim.x >> 5) + wave;
    const int units = mtiles * (N / (16 * NT));
    if (unit >= units) return;
    const int mtile = unit % mtiles;
    const int ngrp  = unit / mtiles;

    const int rlo = lane & 15;
    const int hiA = (lane >> 4) << 3;   // 0 or 8  (A K-half)
    const int hiB = (lane >> 4) << 3;   // 0 or 8  (B K-half)

    int m = mtile * 16 + rlo;
    int srcRow = m;
    if (flipIn) { int b = m / LSEQ, l = m % LSEQ; srcRow = b * LSEQ + (LSEQ - 1 - l); }
    const __bf16* arow = act + (size_t)srcRow * K;

    const __bf16* wrow[NT];
#pragma unroll
    for (int t = 0; t < NT; ++t) {
        const int ncol = (ngrp * NT + t) * 16 + rlo;
        wrow[t] = W + (size_t)ncol * K;
    }

    v8f acc[NT] = {};

    // ---- ping-pong pipelined k loop (K % 64 == 0) ----
    v16bf a0, a1, b0[NT], b1[NT];
    a0 = load_frag_a(arow, 0, hiA);
#pragma unroll
    for (int t = 0; t < NT; ++t) b0[t] = load_frag_b(wrow[t], 0, hiB);

    int k0 = 0;
    for (; k0 + 64 < K; k0 += 64) {
        __builtin_prefetch(arow + k0 + 64, 0, 1);
        a1 = load_frag_a(arow, k0 + 32, hiA);
#pragma unroll
        for (int t = 0; t < NT; ++t) b1[t] = load_frag_b(wrow[t], k0 + 32, hiB);
#pragma unroll
        for (int t = 0; t < NT; ++t)
            acc[t] = __builtin_amdgcn_wmma_f32_16x16x32_bf16(
                false, a0, false, b0[t], (short)0, acc[t], false, false);
        a0 = load_frag_a(arow, k0 + 64, hiA);
#pragma unroll
        for (int t = 0; t < NT; ++t) b0[t] = load_frag_b(wrow[t], k0 + 64, hiB);
#pragma unroll
        for (int t = 0; t < NT; ++t)
            acc[t] = __builtin_amdgcn_wmma_f32_16x16x32_bf16(
                false, a1, false, b1[t], (short)0, acc[t], false, false);
    }
    // tail: exactly 64 k remaining
    a1 = load_frag_a(arow, k0 + 32, hiA);
#pragma unroll
    for (int t = 0; t < NT; ++t) b1[t] = load_frag_b(wrow[t], k0 + 32, hiB);
#pragma unroll
    for (int t = 0; t < NT; ++t)
        acc[t] = __builtin_amdgcn_wmma_f32_16x16x32_bf16(
            false, a0, false, b0[t], (short)0, acc[t], false, false);
#pragma unroll
    for (int t = 0; t < NT; ++t)
        acc[t] = __builtin_amdgcn_wmma_f32_16x16x32_bf16(
            false, a1, false, b1[t], (short)0, acc[t], false, false);

    // D layout: lane 0-15 -> N=lane, VGPR r -> M=r; lane 16-31 -> M=r+8
    const int rowAdd = (lane >> 4) << 3;
#pragma unroll
    for (int t = 0; t < NT; ++t) {
        const int col = (ngrp * NT + t) * 16 + rlo;
        const float bv = bias ? bias[col] : 0.0f;
#pragma unroll
        for (int r = 0; r < 8; ++r) {
            int mo = mtile * 16 + r + rowAdd;
            int dstRow = mo;
            if (flipOut) { int b = mo / LSEQ, l = mo % LSEQ; dstRow = b * LSEQ + (LSEQ - 1 - l); }
            const size_t idx = (size_t)dstRow * outStride + outColOff + col;
            const float val = acc[t][r] + bv;
            if constexpr (OBF) ((__bf16*)outp)[idx] = f2bf(val);
            else               ((float*)outp)[idx]  = val;
        }
    }
}

// bulk f32 -> bf16 converter, 4 elements/thread
__global__ void cvt_f32_bf16(const float* __restrict__ in, __bf16* __restrict__ out,
                             int n4) {
    int i = blockIdx.x * blockDim.x + threadIdx.x;
    if (i >= n4) return;
    float4 q = ((const float4*)in)[i];
    v4bf o;
    o[0] = f2bf(q.x); o[1] = f2bf(q.y); o[2] = f2bf(q.z); o[3] = f2bf(q.w);
    ((v4bf*)out)[i] = o;
}

// depthwise causal conv1d (k=4) + bias + SiLU; xi = first DI columns of xz.
// Emits xc in f32 (scan consumer) and bf16 (x_proj GEMM consumer).
__global__ void conv_silu_kernel(const float* __restrict__ xz,
                                 const float* __restrict__ cw,
                                 const float* __restrict__ cb,
                                 float* __restrict__ xc,
                                 __bf16* __restrict__ xcb) {
    int idx = blockIdx.x * blockDim.x + threadIdx.x;   // over B*L*DI
    if (idx >= BATCH * LSEQ * DI) return;
    int d   = idx & (DI - 1);
    int row = idx >> 9;               // b*L + l
    int l   = row & (LSEQ - 1);
    int base = row - l;               // b*L
    float s = cb[d];
#pragma unroll
    for (int j = 0; j < 4; ++j) {
        int lj = l + j - 3;
        if (lj >= 0) s += xz[(size_t)(base + lj) * (2 * DI) + d] * cw[d * 4 + j];
    }
    s = s / (1.0f + __expf(-s));
    xc[idx]  = s;
    xcb[idx] = f2bf(s);
}

// chunked selective scan, pass 1: per (b,d,chunk) compute softplus(dt) (cached),
// per-chunk sum(dt) and h_end starting from h=0. All row accesses are b128.
__global__ void scan_pass1(const float* __restrict__ dbl, const float* __restrict__ xc,
                           const float* __restrict__ dtw_, const float* __restrict__ dtb_,
                           const float* __restrict__ Alog,
                           float* __restrict__ dtbuf, float* __restrict__ hEnd,
                           float* __restrict__ sumdt) {
    int tid = blockIdx.x * blockDim.x + threadIdx.x;   // (b*DI + d)*NCHUNK + c
    if (tid >= BATCH * DI * NCHUNK) return;
    int c  = tid % NCHUNK;
    int bd = tid / NCHUNK;
    int d  = bd % DI;
    int b  = bd / DI;

    float dtw[RANK], Arow[NSTATE], h[NSTATE];
#pragma unroll
    for (int q = 0; q < RANK / 4; ++q)
        ((float4*)dtw)[q] = ((const float4*)(dtw_ + d * RANK))[q];
#pragma unroll
    for (int q = 0; q < NSTATE / 4; ++q)
        ((float4*)Arow)[q] = ((const float4*)(Alog + d * NSTATE))[q];
#pragma unroll
    for (int n = 0; n < NSTATE; ++n) { Arow[n] = -__expf(Arow[n]); h[n] = 0.0f; }
    const float dtb = dtb_[d];
    float sdt = 0.0f;

    const int l0 = c * LCHUNK;
    for (int l = l0; l < l0 + LCHUNK; ++l) {
        const size_t rowi = (size_t)(b * LSEQ + l);
        float rowv[48];
#pragma unroll
        for (int q = 0; q < 12; ++q)
            ((float4*)rowv)[q] = ((const float4*)(dbl + rowi * 48))[q];
        float dtv = dtb;
#pragma unroll
        for (int r = 0; r < RANK; ++r) dtv += rowv[r] * dtw[r];
        dtv = (dtv > 20.0f) ? dtv : logf(1.0f + __expf(dtv));
        dtbuf[rowi * DI + d] = dtv;
        sdt += dtv;
        const float xcv = xc[rowi * DI + d];
#pragma unroll
        for (int n = 0; n < NSTATE; ++n)
            h[n] = __expf(dtv * Arow[n]) * h[n] + dtv * rowv[16 + n] * xcv;
    }
#pragma unroll
    for (int q = 0; q < NSTATE / 4; ++q)
        ((float4*)(hEnd + (size_t)tid * NSTATE))[q] = ((float4*)h)[q];
    sumdt[tid] = sdt;
}

// prefix across chunks per (b,d,n): chunk transfer factor = exp(A[n]*sum_dt)
__global__ void scan_prefix(const float* __restrict__ Alog,
                            const float* __restrict__ sumdt,
                            const float* __restrict__ hEnd,
                            float* __restrict__ hInit) {
    int tid = blockIdx.x * blockDim.x + threadIdx.x;   // (b*DI + d)*NSTATE + n
    if (tid >= BATCH * DI * NSTATE) return;
    int n  = tid % NSTATE;
    int bd = tid / NSTATE;
    int d  = bd % DI;
    const float An = -__expf(Alog[d * NSTATE + n]);
    float carry = 0.0f;
    for (int c = 0; c < NCHUNK; ++c) {
        const size_t idx = (size_t)bd * NCHUNK + c;
        hInit[idx * NSTATE + n] = carry;
        carry = __expf(An * sumdt[idx]) * carry + hEnd[idx * NSTATE + n];
    }
}

// pass 2: replay chunk from correct initial state, emit y = (h.C + D*xc)*silu(z)
// y written in bf16 (only consumed by the out_proj GEMM).
__global__ void scan_pass2(const float* __restrict__ dbl, const float* __restrict__ xc,
                           const float* __restrict__ xz, const float* __restrict__ dtbuf,
                           const float* __restrict__ Alog, const float* __restrict__ Dp,
                           const float* __restrict__ hInit, __bf16* __restrict__ y) {
    int tid = blockIdx.x * blockDim.x + threadIdx.x;   // (b*DI + d)*NCHUNK + c
    if (tid >= BATCH * DI * NCHUNK) return;
    int c  = tid % NCHUNK;
    int bd = tid / NCHUNK;
    int d  = bd % DI;
    int b  = bd / DI;

    float Arow[NSTATE], h[NSTATE];
#pragma unroll
    for (int q = 0; q < NSTATE / 4; ++q) {
        ((float4*)Arow)[q] = ((const float4*)(Alog + d * NSTATE))[q];
        ((float4*)h)[q]    = ((const float4*)(hInit + (size_t)tid * NSTATE))[q];
    }
#pragma unroll
    for (int n = 0; n < NSTATE; ++n) Arow[n] = -__expf(Arow[n]);
    const float Dv = Dp[d];

    const int l0 = c * LCHUNK;
    for (int l = l0; l < l0 + LCHUNK; ++l) {
        const size_t rowi = (size_t)(b * LSEQ + l);
        float rowv[48];
#pragma unroll
        for (int q = 0; q < 12; ++q)
            ((float4*)rowv)[q] = ((const float4*)(dbl + rowi * 48))[q];
        const float dtv = dtbuf[rowi * DI + d];
        const float xcv = xc[rowi * DI + d];
        float acc = 0.0f;
#pragma unroll
        for (int n = 0; n < NSTATE; ++n) {
            h[n] = __expf(dtv * Arow[n]) * h[n] + dtv * rowv[16 + n] * xcv;
            acc += h[n] * rowv[32 + n];
        }
        const float zv = xz[rowi * (2 * DI) + DI + d];
        const float yv = acc + Dv * xcv;
        y[rowi * DI + d] = f2bf(yv * (zv / (1.0f + __expf(-zv))));
    }
}

// ---------------------------------------------------------------------------
extern "C" void kernel_launch(void* const* d_in, const int* in_sizes, int n_in,
                              void* d_out, int out_size, void* d_ws, size_t ws_size,
                              hipStream_t stream) {
    (void)in_sizes; (void)n_in; (void)out_size; (void)ws_size;
    const float* x       = (const float*)d_in[0];
    const float* mixer_w = (const float*)d_in[1];
    const float* mixer_b = (const float*)d_in[2];

    struct Dir {
        const float *in_w, *conv_w, *conv_b, *xproj_w, *dt_w, *dt_b, *A_log, *Dp, *out_w;
    };
    Dir fw = { (const float*)d_in[3],  (const float*)d_in[4],  (const float*)d_in[5],
               (const float*)d_in[6],  (const float*)d_in[7],  (const float*)d_in[8],
               (const float*)d_in[9],  (const float*)d_in[10], (const float*)d_in[11] };
    Dir bw = { (const float*)d_in[12], (const float*)d_in[13], (const float*)d_in[14],
               (const float*)d_in[15], (const float*)d_in[16], (const float*)d_in[17],
               (const float*)d_in[18], (const float*)d_in[19], (const float*)d_in[20] };

    constexpr size_t MB = 1ull << 20;
    char* ws = (char*)d_ws;
    // f32 buffers
    float*  xz_fw    = (float*) (ws +   0 * MB);  // 16 MB : (B,L,1024)
    float*  xz_bw    = (float*) (ws +  16 * MB);  // 16 MB
    float*  xc_fw    = (float*) (ws +  32 * MB);  //  8 MB : (B,L,512) f32 (scan)
    float*  xc_bw    = (float*) (ws +  40 * MB);  //  8 MB
    float*  dbl_fw   = (float*) (ws +  48 * MB);  // .75MB : (B,L,48)
    float*  dbl_bw   = (float*) (ws +  49 * MB);
    float*  dt_fw    = (float*) (ws +  50 * MB);  //  8 MB : softplus(dt)
    float*  dt_bw    = (float*) (ws +  58 * MB);
    float*  hEnd_fw  = (float*) (ws +  66 * MB);  //  1 MB : (B,DI,NCHUNK,16)
    float*  hEnd_bw  = (float*) (ws +  67 * MB);
    float*  hInit_fw = (float*) (ws +  68 * MB);
    float*  hInit_bw = (float*) (ws +  69 * MB);
    float*  sdt_fw   = (float*) (ws +  70 * MB);  // 64 KB
    float*  sdt_bw   = (float*) (ws +  71 * MB);
    // bf16 buffers
    __bf16* x_bf     = (__bf16*)(ws +  72 * MB);  //  2 MB : x in bf16
    __bf16* xcb_fw   = (__bf16*)(ws +  74 * MB);  //  4 MB : xc in bf16
    __bf16* xcb_bw   = (__bf16*)(ws +  78 * MB);
    __bf16* y_fw     = (__bf16*)(ws +  82 * MB);  //  4 MB : gated scan output
    __bf16* y_bw     = (__bf16*)(ws +  86 * MB);
    __bf16* comb     = (__bf16*)(ws +  90 * MB);  //  4 MB : (B,L,512) fwd|bwd
    __bf16* w_in_fw  = (__bf16*)(ws +  94 * MB);  // 0.5MB : (1024,256)
    __bf16* w_in_bw  = (__bf16*)(ws +  95 * MB);
    __bf16* w_xp_fw  = (__bf16*)(ws +  96 * MB);  //  48KB : (48,512)
    __bf16* w_xp_bw  = (__bf16*)(ws +  97 * MB);
    __bf16* w_out_fw = (__bf16*)(ws +  98 * MB);  // 256KB : (256,512)
    __bf16* w_out_bw = (__bf16*)(ws +  99 * MB);
    __bf16* w_mx     = (__bf16*)(ws + 100 * MB);  // 256KB : (256,512)

    const int M = MROWS;                                // 4096
    const int elemBlocks = (BATCH * LSEQ * DI) / 256;   // 8192
    const int scanBlocks = (BATCH * DI * NCHUNK) / 256; // 64
    const int prefBlocks = (BATCH * DI * NSTATE) / 256; // 128

    auto cvt = [&](const float* src, __bf16* dst, int n) {
        cvt_f32_bf16<<<(n / 4 + 255) / 256, 256, 0, stream>>>(src, dst, n / 4);
    };
    // one-time precision conversions (bandwidth-bound, tiny)
    cvt(x,          x_bf,     MROWS * DMODEL);
    cvt(fw.in_w,    w_in_fw,  2 * DI * DMODEL);
    cvt(bw.in_w,    w_in_bw,  2 * DI * DMODEL);
    cvt(fw.xproj_w, w_xp_fw,  48 * DI);
    cvt(bw.xproj_w, w_xp_bw,  48 * DI);
    cvt(fw.out_w,   w_out_fw, DMODEL * DI);
    cvt(bw.out_w,   w_out_bw, DMODEL * DI);
    cvt(mixer_w,    w_mx,     DMODEL * 2 * DMODEL);

    auto runDir = [&](const Dir& P, const __bf16* w_in, const __bf16* w_xp,
                      float* xz, float* xcf, __bf16* xcb, float* dblb, float* dtb,
                      float* hE, float* hI, float* sd, __bf16* yb, int flip) {
        // in_proj: (4096 x 1024), K=256; 16x64 tiles -> 4096 units -> 512 blocks
        gemm_bf16_wmma<4, false><<<512, 256, 0, stream>>>(x_bf, w_in, nullptr, xz,
                                                          M, 2 * DI, DMODEL, flip, 0,
                                                          2 * DI, 0);
        conv_silu_kernel<<<elemBlocks, 256, 0, stream>>>(xz, P.conv_w, P.conv_b, xcf, xcb);
        // x_proj: (4096 x 48), K=512; 16x48 tiles -> 256 units -> 32 blocks
        gemm_bf16_wmma<3, false><<<32, 256, 0, stream>>>(xcb, w_xp, nullptr, dblb,
                                                         M, 48, DI, 0, 0, 48, 0);
        scan_pass1<<<scanBlocks, 256, 0, stream>>>(dblb, xcf, P.dt_w, P.dt_b, P.A_log,
                                                   dtb, hE, sd);
        scan_prefix<<<prefBlocks, 256, 0, stream>>>(P.A_log, sd, hE, hI);
        scan_pass2<<<scanBlocks, 256, 0, stream>>>(dblb, xcf, xz, dtb, P.A_log, P.Dp,
                                                   hI, yb);
    };

    runDir(fw, w_in_fw, w_xp_fw, xz_fw, xc_fw, xcb_fw, dbl_fw, dt_fw,
           hEnd_fw, hInit_fw, sdt_fw, y_fw, 0);
    runDir(bw, w_in_bw, w_xp_bw, xz_bw, xc_bw, xcb_bw, dbl_bw, dt_bw,
           hEnd_bw, hInit_bw, sdt_bw, y_bw, 1);

    // out_proj into combined bf16 (fwd -> cols 0..255, bwd -> cols 256..511 un-flipped)
    gemm_bf16_wmma<4, true><<<128, 256, 0, stream>>>(y_fw, w_out_fw, nullptr, comb,
                                                     M, DMODEL, DI, 0, 0,
                                                     2 * DMODEL, 0);
    gemm_bf16_wmma<4, true><<<128, 256, 0, stream>>>(y_bw, w_out_bw, nullptr, comb,
                                                     M, DMODEL, DI, 0, 1,
                                                     2 * DMODEL, DMODEL);
    // mixer: (4096 x 256) = comb (4096 x 512) @ mixer_w^T + mixer_b -> f32 output
    gemm_bf16_wmma<4, false><<<128, 256, 0, stream>>>(comb, w_mx, mixer_b,
                                                      (float*)d_out,
                                                      M, DMODEL, 2 * DMODEL, 0, 0,
                                                      DMODEL, 0);
}
